// GAT_86887188399073
// MI455X (gfx1250) — matrix operand order repaired and checked by reading, not compiled, not run on hardware
//
#include <hip/hip_runtime.h>
#include <cstdint>
#include <cstddef>

// ---------------------------------------------------------------------------
// GAT (2-layer, PyG-style) for MI455X / gfx1250.
// GEMMs via v_wmma_f32_16x16x32_f16 with LDS-staged f16 tiles (fragments are
// pure ds_load_b128). Edge softmax: 3-pass (atomic max via order-preserving
// uint encoding, atomic sum, normalize+scatter-add, 8 channels per thread).
// Whole working set (~157MB) fits in the 192MB global L2.
// ---------------------------------------------------------------------------

typedef __attribute__((ext_vector_type(16))) _Float16 v16h;
typedef __attribute__((ext_vector_type(8)))  float    v8f;

#define NEG_SLOPE 0.2f
#define EPS_DEN   1e-16f

__device__ __forceinline__ float lrelu(float v) { return v > 0.f ? v : NEG_SLOPE * v; }

// Order-preserving float<->uint mapping (monotonic) for atomicMax on floats.
__device__ __forceinline__ unsigned enc_f(float f) {
  unsigned u = __float_as_uint(f);
  return (u & 0x80000000u) ? ~u : (u | 0x80000000u);
}
__device__ __forceinline__ float dec_f(unsigned u) {
  u = (u & 0x80000000u) ? (u & 0x7FFFFFFFu) : ~u;
  return __uint_as_float(u);
}

// ---------------------------------------------------------------------------
// WMMA GEMM: H = X[nrows x K] @ W[K x NC], f32 in/out, f16 WMMA math.
// Block = 128 threads = 4 waves, 64 rows per block, NT column tiles of 16.
// X tile (64 x K) and W (transposed, [NT*16][K]) staged in LDS as f16;
// WMMA fragments then load as two ds_load_b128 each.
// ---------------------------------------------------------------------------
template<int K, int NC, int NT>
__global__ void gemm_wmma_kernel(const float* __restrict__ X,
                                 const float* __restrict__ W,
                                 float* __restrict__ Hout, int nrows) {
  constexpr int NTC = NT * 16;                   // padded output cols
  __shared__ __align__(16) _Float16 Xs[64 * K];  // [row][k]
  __shared__ __align__(16) _Float16 Wt[NTC * K]; // [n][k]  (transposed)

  const int tid  = threadIdx.x;
  const int wave = tid >> 5;
  const int lane = tid & 31;
  const int hi   = lane >> 4;      // half-wave: 0 (lanes 0-15), 1 (lanes 16-31)
  const int lo   = lane & 15;
  const int rblk = blockIdx.x * 64;

  // ---- Stage X tile (64 rows x K) as f16, packed pair stores ----
  for (int p = tid; p < 64 * (K / 2); p += 128) {
    const int lr = p / (K / 2);
    const int kk = (p % (K / 2)) * 2;
    int r = rblk + lr;
    if (r >= nrows) r = nrows - 1;               // clamp (stores are guarded)
    const float2 f = *reinterpret_cast<const float2*>(X + (size_t)r * K + kk);
    union { _Float16 h[2]; unsigned u; } pk;
    pk.h[0] = (_Float16)f.x;
    pk.h[1] = (_Float16)f.y;
    *reinterpret_cast<unsigned*>(&Xs[lr * K + kk]) = pk.u;
  }
  // ---- Stage W transposed ([n][k]) as f16, zero-pad cols n >= NC ----
  for (int p = tid; p < NTC * (K / 2); p += 128) {
    const int n  = p / (K / 2);
    const int kk = (p % (K / 2)) * 2;
    union { _Float16 h[2]; unsigned u; } pk;
    if (n < NC) {
      pk.h[0] = (_Float16)W[(size_t)kk * NC + n];
      pk.h[1] = (_Float16)W[(size_t)(kk + 1) * NC + n];
    } else {
      pk.u = 0u;
    }
    *reinterpret_cast<unsigned*>(&Wt[n * K + kk]) = pk.u;
  }
  __syncthreads();

  const int lr0 = wave * 16;                     // wave's 16-row tile in LDS
  v8f acc[NT] = {};

  for (int kb = 0; kb < K; kb += 32) {
    // A fragment: lane row = lo; halves v0..3 -> k = kb+hi*8+[0,8),
    //             v4..7 -> k = kb+16+hi*8+[0,8). Two b128 LDS reads.
    v16h a;
    {
      const uint4* p0 = reinterpret_cast<const uint4*>(&Xs[(lr0 + lo) * K + kb + hi * 8]);
      const uint4* p1 = reinterpret_cast<const uint4*>(&Xs[(lr0 + lo) * K + kb + 16 + hi * 8]);
      reinterpret_cast<uint4*>(&a)[0] = *p0;
      reinterpret_cast<uint4*>(&a)[1] = *p1;
    }
#pragma unroll
    for (int t = 0; t < NT; ++t) {
      // B fragment: lane col n = t*16+lo; k = kb+hi*8+[0,8), then kb+16+hi*8+[0,8) (matches A).
      v16h b;
      const int n = t * 16 + lo;
      const uint4* q0 = reinterpret_cast<const uint4*>(&Wt[n * K + kb + hi * 8]);
      const uint4* q1 = reinterpret_cast<const uint4*>(&Wt[n * K + kb + 16 + hi * 8]);
      reinterpret_cast<uint4*>(&b)[0] = *q0;
      reinterpret_cast<uint4*>(&b)[1] = *q1;
      acc[t] = __builtin_amdgcn_wmma_f32_16x16x32_f16(
          false, a, false, b, (short)0, acc[t], false, false);
    }
  }

  // ---- Store C/D: vgpr v -> (M = v + hi*8, N = lane&15) ----
#pragma unroll
  for (int t = 0; t < NT; ++t) {
    const int n = t * 16 + lo;
    if (n < NC) {
#pragma unroll
      for (int v = 0; v < 8; ++v) {
        const int r = rblk + lr0 + v + hi * 8;
        if (r < nrows) Hout[(size_t)r * NC + n] = acc[t][v];
      }
    }
  }
}

// ---------------------------------------------------------------------------
// Per-node attention logits: As[n,h] = sum_c H[n,h,c]*a_s[h,c]  (same for Ad)
// ---------------------------------------------------------------------------
template<int H, int C>
__global__ void alpha_kernel(const float* __restrict__ Hm,
                             const float* __restrict__ a_s,
                             const float* __restrict__ a_d,
                             float* __restrict__ As, float* __restrict__ Ad,
                             int n) {
  const int idx = blockIdx.x * blockDim.x + threadIdx.x;   // (node, head)
  if (idx >= n * H) return;
  const int node = idx / H, h = idx % H;
  const float* hp = Hm + (size_t)node * (H * C) + h * C;
  float s = 0.f, d = 0.f;
#pragma unroll
  for (int c = 0; c < C; ++c) {
    const float v = hp[c];
    s += v * a_s[h * C + c];
    d += v * a_d[h * C + c];
  }
  As[idx] = s;
  Ad[idx] = d;
}

// ---------------------------------------------------------------------------
// Fill helpers (buffers must be re-initialized every call)
// ---------------------------------------------------------------------------
__global__ void fill_f32(float* __restrict__ p, float v, int n) {
  const int i = blockIdx.x * blockDim.x + threadIdx.x;
  if (i < n) p[i] = v;
}
__global__ void fill_u32(unsigned* __restrict__ p, unsigned v, int n) {
  const int i = blockIdx.x * blockDim.x + threadIdx.x;
  if (i < n) p[i] = v;
}

__device__ __forceinline__ void edge_endpoints(const long long* __restrict__ ei,
                                               int E, int e, int& src, int& dst) {
  if (e < E) {
    src = (int)ei[e];
    dst = (int)ei[(size_t)E + e];
  } else {            // self loops appended after the E real edges
    src = dst = e - E;
  }
}

// ---------------------------------------------------------------------------
// Pass 1: segment max of leaky_relu(As[src]+Ad[dst]) per destination node
// ---------------------------------------------------------------------------
template<int H>
__global__ void edge_max_kernel(const long long* __restrict__ ei, int E, int n,
                                const float* __restrict__ As,
                                const float* __restrict__ Ad,
                                unsigned* __restrict__ Menc) {
  const int e = blockIdx.x * blockDim.x + threadIdx.x;
  if (e >= E + n) return;
  int src, dst;
  edge_endpoints(ei, E, e, src, dst);
#pragma unroll
  for (int h = 0; h < H; ++h) {
    const float v = lrelu(As[src * H + h] + Ad[dst * H + h]);
    atomicMax(&Menc[dst * H + h], enc_f(v));
  }
}

// ---------------------------------------------------------------------------
// Pass 2: segment sum of exp(v - m[dst]) per destination node
// ---------------------------------------------------------------------------
template<int H>
__global__ void edge_sum_kernel(const long long* __restrict__ ei, int E, int n,
                                const float* __restrict__ As,
                                const float* __restrict__ Ad,
                                const unsigned* __restrict__ Menc,
                                float* __restrict__ Den) {
  const int e = blockIdx.x * blockDim.x + threadIdx.x;
  if (e >= E + n) return;
  int src, dst;
  edge_endpoints(ei, E, e, src, dst);
#pragma unroll
  for (int h = 0; h < H; ++h) {
    const float v = lrelu(As[src * H + h] + Ad[dst * H + h]);
    const float m = dec_f(Menc[dst * H + h]);
    atomicAdd(&Den[dst * H + h], __expf(v - m));
  }
}

// ---------------------------------------------------------------------------
// Pass 3: normalize + scatter-add messages.
// One thread per (edge, CG-channel group); alpha computed once per group.
// CG must divide C (layer1: C=8) or the group must sit inside head 0
// (layer2: H=1, C=40, CG=8 -> 5 groups, all head 0).
// ---------------------------------------------------------------------------
template<int H, int C, int CG>
__global__ void edge_scatter_kernel(const long long* __restrict__ ei, int E, int n,
                                    const float* __restrict__ As,
                                    const float* __restrict__ Ad,
                                    const unsigned* __restrict__ Menc,
                                    const float* __restrict__ Den,
                                    const float* __restrict__ Hm,
                                    float* __restrict__ Acc) {
  constexpr int CT = H * C;
  constexpr int G  = CT / CG;      // channel groups per edge
  const int idx = blockIdx.x * blockDim.x + threadIdx.x;
  if (idx >= (E + n) * G) return;
  const int e  = idx / G;
  const int g  = idx % G;
  const int c0 = g * CG;
  const int h  = c0 / C;
  int src, dst;
  edge_endpoints(ei, E, e, src, dst);
  const float v = lrelu(As[src * H + h] + Ad[dst * H + h]);
  const float m = dec_f(Menc[dst * H + h]);
  const float alpha = __expf(v - m) / (Den[dst * H + h] + EPS_DEN);
  const float* hp = Hm + (size_t)src * CT + c0;
  float* ap = Acc + (size_t)dst * CT + c0;
#pragma unroll
  for (int c = 0; c < CG; ++c) atomicAdd(&ap[c], hp[c] * alpha);
}

// ---------------------------------------------------------------------------
// out = elu(acc + bias)
// ---------------------------------------------------------------------------
__global__ void bias_elu_kernel(const float* __restrict__ Acc,
                                const float* __restrict__ b,
                                float* __restrict__ Out, int n, int CT) {
  const int idx = blockIdx.x * blockDim.x + threadIdx.x;
  if (idx >= n * CT) return;
  const int c = idx % CT;
  const float v = Acc[idx] + b[c];
  Out[idx] = v > 0.f ? v : expm1f(v);
}

// ---------------------------------------------------------------------------
// Host-side orchestration
// ---------------------------------------------------------------------------
static inline unsigned cdiv(long long a, long long b) {
  return (unsigned)((a + b - 1) / b);
}

extern "C" void kernel_launch(void* const* d_in, const int* in_sizes, int n_in,
                              void* d_out, int out_size, void* d_ws, size_t ws_size,
                              hipStream_t stream) {
  const int F_IN = 128, H1 = 8, C1 = 8, CT1 = H1 * C1;   // 64
  const int NC2 = 40;                                     // classes (1 head)

  const int N = in_sizes[0] / F_IN;    // 100000
  const int E = in_sizes[1] / 2;       // 1600000
  const int Etot = E + N;

  const float*     x      = (const float*)d_in[0];
  const long long* ei     = (const long long*)d_in[1];   // int64 edge_index [2,E]
  // d_in[2] = edge_attr (unused by reference GATConv)
  const float*     W1     = (const float*)d_in[3];
  const float*     a_src1 = (const float*)d_in[4];
  const float*     a_dst1 = (const float*)d_in[5];
  const float*     b1     = (const float*)d_in[6];
  const float*     W2     = (const float*)d_in[7];
  const float*     a_src2 = (const float*)d_in[8];
  const float*     a_dst2 = (const float*)d_in[9];
  const float*     b2     = (const float*)d_in[10];
  float*           out    = (float*)d_out;

  // Workspace layout (floats); total 200*N floats = 80 MB
  float* ws   = (float*)d_ws;
  float* h1   = ws;                    // [N,64]  h of layer1; later reused as x1
  float* acc1 = h1 + (size_t)N * CT1;  // [N,64]  layer1 scatter accum; later h2
  float* As   = acc1 + (size_t)N * CT1;// [N,8]   (layer2 reuses [N,1])
  float* Ad   = As + (size_t)N * H1;   // [N,8]
  unsigned* Menc = (unsigned*)(Ad + (size_t)N * H1); // [N,8]
  float* Den  = (float*)(Menc + (size_t)N * H1);     // [N,8]
  float* acc2 = Den + (size_t)N * H1;  // [N,40]

  const int TB = 256;

  // ================= Layer 1 =================
  // h1 = x @ W1  (K=128, NC=64, 4 col tiles)
  gemm_wmma_kernel<128, 64, 4><<<cdiv(N, 64), 128, 0, stream>>>(x, W1, h1, N);

  alpha_kernel<H1, C1><<<cdiv((long long)N * H1, TB), TB, 0, stream>>>(
      h1, a_src1, a_dst1, As, Ad, N);

  fill_u32<<<cdiv((long long)N * H1, TB), TB, 0, stream>>>(Menc, 0u, N * H1);
  fill_f32<<<cdiv((long long)N * H1, TB), TB, 0, stream>>>(Den, 0.f, N * H1);
  fill_f32<<<cdiv((long long)N * CT1, TB), TB, 0, stream>>>(acc1, 0.f, N * CT1);

  edge_max_kernel<H1><<<cdiv(Etot, TB), TB, 0, stream>>>(ei, E, N, As, Ad, Menc);
  edge_sum_kernel<H1><<<cdiv(Etot, TB), TB, 0, stream>>>(ei, E, N, As, Ad, Menc, Den);
  edge_scatter_kernel<H1, C1, 8><<<cdiv((long long)Etot * (CT1 / 8), TB), TB, 0, stream>>>(
      ei, E, N, As, Ad, Menc, Den, h1, acc1);

  // x1 = elu(acc1 + b1) -> stored into h1's buffer (h1 no longer needed)
  bias_elu_kernel<<<cdiv((long long)N * CT1, TB), TB, 0, stream>>>(acc1, b1, h1, N, CT1);

  // ================= Layer 2 =================
  // h2 = x1 @ W2  (K=64, NC=40, 3 col tiles) -> into acc1's buffer
  float* x1 = h1;
  float* h2 = acc1;
  gemm_wmma_kernel<64, 40, 3><<<cdiv(N, 64), 128, 0, stream>>>(x1, W2, h2, N);

  alpha_kernel<1, NC2><<<cdiv(N, TB), TB, 0, stream>>>(h2, a_src2, a_dst2, As, Ad, N);

  fill_u32<<<cdiv(N, TB), TB, 0, stream>>>(Menc, 0u, N);
  fill_f32<<<cdiv(N, TB), TB, 0, stream>>>(Den, 0.f, N);
  fill_f32<<<cdiv((long long)N * NC2, TB), TB, 0, stream>>>(acc2, 0.f, N * NC2);

  edge_max_kernel<1><<<cdiv(Etot, TB), TB, 0, stream>>>(ei, E, N, As, Ad, Menc);
  edge_sum_kernel<1><<<cdiv(Etot, TB), TB, 0, stream>>>(ei, E, N, As, Ad, Menc, Den);
  edge_scatter_kernel<1, NC2, 8><<<cdiv((long long)Etot * (NC2 / 8), TB), TB, 0, stream>>>(
      ei, E, N, As, Ad, Menc, Den, h2, acc2);

  // out = elu(acc2 + b2)
  bias_elu_kernel<<<cdiv((long long)N * NC2, TB), TB, 0, stream>>>(acc2, b2, out, N, NC2);

  (void)n_in; (void)out_size; (void)ws_size;
}
